// MultiHeadSelfAttention_32074815766859
// MI455X (gfx1250) — matrix operand order repaired and checked by reading, not compiled, not run on hardware
//
#include <hip/hip_runtime.h>
#include <hip/hip_bf16.h>

#define D_MODEL 2048
#define NHEADS  16
#define DHEAD   128
#define BATCH   2
#define SEQ     2048
#define BSROWS  (BATCH * SEQ)   /* 4096 */

typedef unsigned short u16;
typedef __attribute__((ext_vector_type(16))) __bf16 v16bf;
typedef __attribute__((ext_vector_type(8)))  __bf16 v8bf;
typedef __attribute__((ext_vector_type(8)))  float  v8f;
typedef __attribute__((ext_vector_type(4)))  int    v4i;
typedef __attribute__((address_space(1))) v4i g_v4i;   // global int4
typedef __attribute__((address_space(3))) v4i l_v4i;   // LDS int4

// ---------------------------------------------------------------- helpers
__device__ __forceinline__ u16 to_bf16(float f) {
    union { float f; unsigned u; } v; v.f = f;
    unsigned r = v.u + 0x7FFFu + ((v.u >> 16) & 1u);   // round-to-nearest-even
    return (u16)(r >> 16);
}

// Build a 16x32 bf16 A-fragment (ISA 7.12.2): lane L (0-15) holds row M=L,
// K = 0..15; lane L+16 holds row M=L, K = 16..31 (contiguous k order).
__device__ __forceinline__ v16bf ld_afrag(const u16* lo, const u16* hi) {
    v8bf a = *reinterpret_cast<const v8bf*>(lo);
    v8bf b = *reinterpret_cast<const v8bf*>(hi);
    v16bf r;
#pragma unroll
    for (int i = 0; i < 8; ++i) { r[i] = a[i]; r[i + 8] = b[i]; }
    return r;
}

__device__ __forceinline__ v8f wmma_bf16(v16bf a, v16bf b, v8f c) {
    return __builtin_amdgcn_wmma_f32_16x16x32_bf16(
        /*neg_a=*/false, a, /*neg_b=*/false, b,
        /*c_mod=*/(short)0, c, /*reuse_a=*/false, /*reuse_b=*/false);
}

// 16-byte async global->LDS copy (CDNA5 GLOBAL_LOAD_ASYNC_TO_LDS_B128).
// Signature per toolchain: (v4i AS1* src, v4i AS3* dst, imm offset, imm cpol).
__device__ __forceinline__ void async_copy_b128(const u16* gsrc, u16* lds_dst) {
#if __has_builtin(__builtin_amdgcn_global_load_async_to_lds_b128)
    __builtin_amdgcn_global_load_async_to_lds_b128(
        (g_v4i*)gsrc, (l_v4i*)lds_dst, /*imm offset=*/0, /*cpol=*/0);
#else
    *reinterpret_cast<uint4*>(lds_dst) = *reinterpret_cast<const uint4*>(gsrc);
#endif
}

__device__ __forceinline__ void wait_async0() {
#if __has_builtin(__builtin_amdgcn_s_wait_asynccnt)
    __builtin_amdgcn_s_wait_asynccnt(0);
#else
    asm volatile("s_wait_asynccnt 0" ::: "memory");
#endif
}

// ------------------------------------------------ 1. weight transpose f32->bf16
__global__ void wt_transpose_kernel(const float* __restrict__ W, u16* __restrict__ Wt) {
    __shared__ float tile[32][33];
    const int bx = blockIdx.x * 32;   // n
    const int by = blockIdx.y * 32;   // k
    const int tx = threadIdx.x;       // 0..31
    const int ty = threadIdx.y;       // 0..7
#pragma unroll
    for (int i = 0; i < 32; i += 8)
        tile[ty + i][tx] = W[(size_t)(by + ty + i) * D_MODEL + bx + tx];
    __syncthreads();
#pragma unroll
    for (int i = 0; i < 32; i += 8)
        Wt[(size_t)(bx + ty + i) * D_MODEL + by + tx] = to_bf16(tile[tx][ty + i]);
}

// ------------------------------------------------ 2. LayerNorm -> bf16
__global__ void ln_kernel(const float* __restrict__ X, const float* __restrict__ g,
                          const float* __restrict__ bvec, u16* __restrict__ Xn) {
    const int row = blockIdx.x;
    const int tid = threadIdx.x;      // 256 threads
    const float* x = X + (size_t)row * D_MODEL;
    float v[8], s = 0.f, sq = 0.f;
#pragma unroll
    for (int i = 0; i < 8; ++i) {
        v[i] = x[tid + i * 256];
        s += v[i]; sq += v[i] * v[i];
    }
#pragma unroll
    for (int off = 16; off; off >>= 1) {
        s  += __shfl_xor(s, off, 32);
        sq += __shfl_xor(sq, off, 32);
    }
    __shared__ float sh[16];
    const int wid = tid >> 5;
    if ((tid & 31) == 0) { sh[wid] = s; sh[8 + wid] = sq; }
    __syncthreads();
    float ts = 0.f, tq = 0.f;
#pragma unroll
    for (int w = 0; w < 8; ++w) { ts += sh[w]; tq += sh[8 + w]; }
    const float mu  = ts * (1.0f / D_MODEL);
    const float var = tq * (1.0f / D_MODEL) - mu * mu;
    const float rs  = rsqrtf(var + 1e-5f);
    u16* out = Xn + (size_t)row * D_MODEL;
#pragma unroll
    for (int i = 0; i < 8; ++i) {
        const int idx = tid + i * 256;
        out[idx] = to_bf16((v[i] - mu) * rs * g[idx] + bvec[idx]);
    }
}

// ------------------------------------------------ 3. GEMM  C = A * Bt^T + bias
// A: [M,K] bf16 row-major, Bt: [N,K] bf16 row-major (pre-transposed weights).
// 256 threads (8 waves), 128x128 tile, K step 32, double-buffered async LDS.
template <int MODE>
__global__ __launch_bounds__(256) void
gemm_kernel(const u16* __restrict__ A, const u16* __restrict__ Bt,
            const float* __restrict__ bias, void* __restrict__ Cout,
            int M, int N, int Kd) {
    __shared__ alignas(32) u16 As[2][128 * 32];
    __shared__ alignas(32) u16 Bs[2][128 * 32];
    const int tid  = threadIdx.x;
    const int lane = tid & 31, wave = tid >> 5;
    const int l16  = lane & 15, hl = lane >> 4;
    const int n0 = blockIdx.x * 128, m0 = blockIdx.y * 128;
    const int crow = tid >> 1;          // 0..127
    const int ccol = (tid & 1) * 16;    // 0 or 16

    const u16* agp = &A[(size_t)(m0 + crow) * Kd + ccol];   // this thread's rows
    const u16* bgp = &Bt[(size_t)(n0 + crow) * Kd + ccol];
    const int loff = crow * 32 + ccol;

    v8f acc[8] = {};

    // prologue: stage k0 = 0 into buffer 0
    async_copy_b128(agp,     &As[0][loff]);
    async_copy_b128(agp + 8, &As[0][loff + 8]);
    async_copy_b128(bgp,     &Bs[0][loff]);
    async_copy_b128(bgp + 8, &Bs[0][loff + 8]);
    wait_async0();
    __syncthreads();

    int buf = 0;
    for (int k0 = 0; k0 < Kd; k0 += 32) {
        if (k0 + 32 < Kd) {   // issue async stage of next tile into other buffer
            const u16* an = agp + (k0 + 32);
            const u16* bn = bgp + (k0 + 32);
            async_copy_b128(an,     &As[buf ^ 1][loff]);
            async_copy_b128(an + 8, &As[buf ^ 1][loff + 8]);
            async_copy_b128(bn,     &Bs[buf ^ 1][loff]);
            async_copy_b128(bn + 8, &Bs[buf ^ 1][loff + 8]);
        }
        // compute on current buffer
        const u16* arow = &As[buf][(wave * 16 + l16) * 32];
        const v16bf afrag = ld_afrag(arow + hl * 16, arow + hl * 16 + 8);
#pragma unroll
        for (int j = 0; j < 8; ++j) {
            const v16bf bfrag =
                *reinterpret_cast<const v16bf*>(&Bs[buf][(j * 16 + l16) * 32 + hl * 16]);
            acc[j] = wmma_bf16(afrag, bfrag, acc[j]);
        }
        wait_async0();
        __syncthreads();
        buf ^= 1;
    }
    // epilogue: C layout: VGPR r -> row (r | r+8), lane&15 -> col
#pragma unroll
    for (int j = 0; j < 8; ++j) {
        const int n = n0 + j * 16 + l16;
        const float bv = bias[n];
#pragma unroll
        for (int r = 0; r < 8; ++r) {
            const int m = m0 + wave * 16 + r + hl * 8;
            const float v = acc[j][r] + bv;
            if (MODE == 0)
                ((u16*)Cout)[(size_t)m * N + n] = to_bf16(v);
            else
                ((float*)Cout)[(size_t)m * N + n] = v;
        }
    }
}

// ------------------------------------------------ 4. V -> V^T per head
// V row-major [B*S, D_MODEL] -> Vt [B, H, DHEAD, S]
__global__ void vt_kernel(const u16* __restrict__ V, u16* __restrict__ Vt) {
    const size_t o = (size_t)blockIdx.x * 256 + threadIdx.x;
    const int s = (int)(o % SEQ);
    size_t r = o / SEQ;
    const int d = (int)(r % DHEAD); r /= DHEAD;
    const int h = (int)(r % NHEADS);
    const int bb = (int)(r / NHEADS);
    Vt[o] = V[(size_t)(bb * SEQ + s) * D_MODEL + h * DHEAD + d];
}

// ------------------------------------------------ 5. causal flash attention
// one wave per block; __launch_bounds__(32,1) -> full VGPR budget, no spills
__global__ __launch_bounds__(32, 1) void
attn_kernel(const u16* __restrict__ Q, const u16* __restrict__ K,
            const u16* __restrict__ Vt, u16* __restrict__ O) {
    const int lane = threadIdx.x & 31;
    const int l16 = lane & 15, hl = lane >> 4;
    const int q0 = blockIdx.x * 16;
    const int h  = blockIdx.y;
    const int bb = blockIdx.z;
    const float scale = 0.08838834764831845f;   // 1/sqrt(128)
    const float NEG_INF = -__builtin_inff();

    __shared__ alignas(32) u16 Pl[16 * 32];

    // resident Q fragments (A layout), rows q0..q0+15, Dh chunks of 32
    v16bf qf[4];
    const size_t qbase = (size_t)(bb * SEQ + q0 + l16) * D_MODEL + h * DHEAD;
#pragma unroll
    for (int c = 0; c < 4; ++c) {
        const u16* p = &Q[qbase + c * 32 + hl * 16];
        qf[c] = ld_afrag(p, p + 8);
    }

    v8f acc[8] = {};
    float m_[8], l_[8];
#pragma unroll
    for (int r = 0; r < 8; ++r) { m_[r] = NEG_INF; l_[r] = 0.f; }

#pragma unroll 1
    for (int k0 = 0; k0 <= q0 + 15; k0 += 32) {
        // ---- scores: two 16x16 tiles (keys k0..k0+15 and k0+16..k0+31)
        v8f s0 = {}, s1 = {};
        const size_t kb0 = (size_t)(bb * SEQ + k0 + l16) * D_MODEL + h * DHEAD + hl * 16;
        const size_t kb1 = kb0 + (size_t)16 * D_MODEL;
#pragma unroll
        for (int c = 0; c < 4; ++c) {
            const v16bf kf0 = *reinterpret_cast<const v16bf*>(&K[kb0 + c * 32]);
            const v16bf kf1 = *reinterpret_cast<const v16bf*>(&K[kb1 + c * 32]);
            s0 = wmma_bf16(qf[c], kf0, s0);
            s1 = wmma_bf16(qf[c], kf1, s1);
        }
        // ---- mask + online softmax (row = q index; half-wave reductions)
        float ef[8];
#pragma unroll
        for (int r = 0; r < 8; ++r) {
            const int qrow = q0 + r + hl * 8;
            const int key0 = k0 + l16, key1 = key0 + 16;
            float v0 = (key0 <= qrow) ? s0[r] * scale : NEG_INF;
            float v1 = (key1 <= qrow) ? s1[r] * scale : NEG_INF;
            float rm = fmaxf(v0, v1);
#pragma unroll
            for (int off = 8; off; off >>= 1) rm = fmaxf(rm, __shfl_xor(rm, off, 16));
            const float mn = fmaxf(m_[r], rm);
            const float p0 = __expf(v0 - mn);   // exp(-inf)=0 handles mask
            const float p1 = __expf(v1 - mn);
            float rs = p0 + p1;
#pragma unroll
            for (int off = 8; off; off >>= 1) rs += __shfl_xor(rs, off, 16);
            ef[r] = __expf(m_[r] - mn);         // first iter: exp(-inf)=0
            l_[r] = l_[r] * ef[r] + rs;
            m_[r] = mn;
            const int mrow = r + hl * 8;        // stage P row-major 16x32 bf16
            Pl[mrow * 32 + l16]      = to_bf16(p0);
            Pl[mrow * 32 + 16 + l16] = to_bf16(p1);
        }
        __syncthreads();
        // ---- AV: A = P (16 q x 32 keys), B = Vt tiles (32 keys x 16 d)
        const u16* prow = &Pl[l16 * 32 + hl * 16];
        const v16bf pf = ld_afrag(prow, prow + 8);
#pragma unroll
        for (int j = 0; j < 8; ++j) {
            const size_t vb =
                ((size_t)(bb * NHEADS + h) * DHEAD + j * 16 + l16) * SEQ + k0 + hl * 16;
            const v16bf vf = *reinterpret_cast<const v16bf*>(&Vt[vb]);
#pragma unroll
            for (int r = 0; r < 8; ++r) acc[j][r] *= ef[r];
            acc[j] = wmma_bf16(pf, vf, acc[j]);
        }
        __syncthreads();
    }
    // ---- normalize and write merged-head output [B*S, D_MODEL] bf16
#pragma unroll
    for (int j = 0; j < 8; ++j) {
#pragma unroll
        for (int r = 0; r < 8; ++r) {
            const int q = q0 + r + hl * 8;
            const float v = acc[j][r] / l_[r];
            O[(size_t)(bb * SEQ + q) * D_MODEL + h * DHEAD + j * 16 + l16] = to_bf16(v);
        }
    }
}

// ------------------------------------------------ host
extern "C" void kernel_launch(void* const* d_in, const int* in_sizes, int n_in,
                              void* d_out, int out_size, void* d_ws, size_t ws_size,
                              hipStream_t stream) {
    (void)in_sizes; (void)n_in; (void)out_size; (void)ws_size;
    const float* X    = (const float*)d_in[0];
    const float* ln_g = (const float*)d_in[1];
    const float* ln_b = (const float*)d_in[2];
    const float* Wq   = (const float*)d_in[3];
    const float* bq   = (const float*)d_in[4];
    const float* Wk   = (const float*)d_in[5];
    const float* bk   = (const float*)d_in[6];
    const float* Wv   = (const float*)d_in[7];
    const float* bv   = (const float*)d_in[8];
    const float* Wo   = (const float*)d_in[9];
    const float* bo   = (const float*)d_in[10];

    char* ws = (char*)d_ws;
    const size_t WBYTES = (size_t)D_MODEL * D_MODEL * sizeof(u16);   // 8 MB
    const size_t ABYTES = (size_t)BSROWS * D_MODEL * sizeof(u16);    // 16 MB
    u16* WqT = (u16*)(ws + 0 * WBYTES);
    u16* WkT = (u16*)(ws + 1 * WBYTES);
    u16* WvT = (u16*)(ws + 2 * WBYTES);
    u16* WoT = (u16*)(ws + 3 * WBYTES);
    char* p  = ws + 4 * WBYTES;
    u16* Xn  = (u16*)(p);             p += ABYTES;
    u16* Qb  = (u16*)(p);             p += ABYTES;
    u16* Kb  = (u16*)(p);             p += ABYTES;
    u16* Vb  = (u16*)(p);             p += ABYTES;
    u16* Vt  = (u16*)(p);             p += ABYTES;
    u16* Ob  = (u16*)(p);

    dim3 tb(32, 8), tg(D_MODEL / 32, D_MODEL / 32);
    wt_transpose_kernel<<<tg, tb, 0, stream>>>(Wq, WqT);
    wt_transpose_kernel<<<tg, tb, 0, stream>>>(Wk, WkT);
    wt_transpose_kernel<<<tg, tb, 0, stream>>>(Wv, WvT);
    wt_transpose_kernel<<<tg, tb, 0, stream>>>(Wo, WoT);

    ln_kernel<<<BSROWS, 256, 0, stream>>>(X, ln_g, ln_b, Xn);

    dim3 gg(D_MODEL / 128, BSROWS / 128);
    gemm_kernel<0><<<gg, 256, 0, stream>>>(Xn, WqT, bq, Qb, BSROWS, D_MODEL, D_MODEL);
    gemm_kernel<0><<<gg, 256, 0, stream>>>(Xn, WkT, bk, Kb, BSROWS, D_MODEL, D_MODEL);
    gemm_kernel<0><<<gg, 256, 0, stream>>>(Xn, WvT, bv, Vb, BSROWS, D_MODEL, D_MODEL);

    vt_kernel<<<(BATCH * NHEADS * DHEAD * SEQ) / 256, 256, 0, stream>>>(Vb, Vt);

    dim3 ag(SEQ / 16, NHEADS, BATCH);
    attn_kernel<<<ag, 32, 0, stream>>>(Qb, Kb, Vt, Ob);

    gemm_kernel<1><<<gg, 256, 0, stream>>>(Ob, WoT, bo, d_out, BSROWS, D_MODEL, D_MODEL);
}
